// MoELayer_11948599018063
// MI455X (gfx1250) — matrix operand order repaired and checked by reading, not compiled, not run on hardware
//
#include <hip/hip_runtime.h>
#include <hip/hip_bf16.h>

// ---------------- problem constants (from reference) ----------------
#define Bv 4
#define Nv 2048
#define Cv 1024
#define Fv 4096
#define Ev 8
#define Kv 2
#define Tv (Bv * Nv)            // 8192 tokens
#define TILE_M 128              // block tile (M and N)
#define SLOT_MAX (Tv * Kv + Ev * TILE_M)   // 17408 padded slots

typedef __bf16 bf16_t;
typedef __attribute__((ext_vector_type(16))) __bf16 v16bf;
typedef __attribute__((ext_vector_type(8)))  __bf16 v8bf;
typedef __attribute__((ext_vector_type(4)))  __bf16 v4bf;
typedef __attribute__((ext_vector_type(8)))  float  v8f;
typedef __attribute__((ext_vector_type(4)))  float  v4f;

// ctrl layout (ints): [0..7]=counts, [8..15]=scatter cursors, [16..24]=padded offsets

// ---------------- tiny control / cast kernels ----------------
__global__ void zero_ctrl_kernel(int* __restrict__ ctrl) {
  if (threadIdx.x < 32) ctrl[threadIdx.x] = 0;
}

__global__ __launch_bounds__(256) void cast_bf16_kernel(const float* __restrict__ src,
                                                        bf16_t* __restrict__ dst, int n4) {
  int i = blockIdx.x * blockDim.x + threadIdx.x;
  if (i >= n4) return;
  v4f v = *(const v4f*)(src + (size_t)i * 4);
  v4bf o;
#pragma unroll
  for (int r = 0; r < 4; ++r) o[r] = (bf16_t)v[r];
  *(v4bf*)(dst + (size_t)i * 4) = o;
}

// ---------------- router: logits, softmax, top-2, counts ----------------
__global__ __launch_bounds__(256) void router_kernel(
    const float* __restrict__ x, const float* __restrict__ rw,
    int* __restrict__ ctrl, int* __restrict__ tok_e, float* __restrict__ tok_w) {
  __shared__ float s_rw[Ev * Cv];                       // 32 KB of the 320 KB WGP LDS
  for (int i = threadIdx.x; i < Ev * Cv; i += 256) s_rw[i] = rw[i];
  __syncthreads();

  int t = blockIdx.x * 256 + threadIdx.x;              // grid sized so t < Tv
  const float* xr = x + (size_t)t * Cv;
  float acc[Ev];
#pragma unroll
  for (int e = 0; e < Ev; ++e) acc[e] = 0.f;
  for (int c = 0; c < Cv; c += 4) {
    v4f xv = *(const v4f*)(xr + c);
#pragma unroll
    for (int e = 0; e < Ev; ++e) {
      acc[e] += xv[0] * s_rw[e * Cv + c]     + xv[1] * s_rw[e * Cv + c + 1]
              + xv[2] * s_rw[e * Cv + c + 2] + xv[3] * s_rw[e * Cv + c + 3];
    }
  }
  // top-2 by logits (softmax is monotonic), ties -> lowest index like jax top_k
  int   i0 = 0;  float l0 = acc[0];
  int   i1 = -1; float l1 = -3.4e38f;
#pragma unroll
  for (int e = 1; e < Ev; ++e) {
    if (acc[e] > l0)      { l1 = l0; i1 = i0; l0 = acc[e]; i0 = e; }
    else if (acc[e] > l1) { l1 = acc[e]; i1 = e; }
  }
  float m = l0;                                        // max logit
  float s = 0.f;
#pragma unroll
  for (int e = 0; e < Ev; ++e) s += expf(acc[e] - m);
  float p0 = expf(l0 - m) / s;
  float p1 = expf(l1 - m) / s;
  float inv = 1.f / (p0 + p1 + 1e-9f);
  tok_e[t * 2]     = i0;  tok_e[t * 2 + 1] = i1;
  tok_w[t * 2]     = p0 * inv;
  tok_w[t * 2 + 1] = p1 * inv;
  atomicAdd(&ctrl[i0], 1);
  atomicAdd(&ctrl[i1], 1);
}

// ---------------- prefix scan (E=8, 128-padded regions) ----------------
__global__ void scan_kernel(int* __restrict__ ctrl) {
  if (threadIdx.x == 0 && blockIdx.x == 0) {
    int off = 0;
    for (int e = 0; e < Ev; ++e) {
      ctrl[16 + e] = off;
      off += (ctrl[e] + TILE_M - 1) & ~(TILE_M - 1);
    }
    ctrl[16 + Ev] = off;
  }
}

// ---------------- scatter tokens into per-expert slot regions ----------------
__global__ __launch_bounds__(256) void scatter_kernel(
    const int* __restrict__ tok_e, int* __restrict__ ctrl,
    int* __restrict__ slot_tok, int* __restrict__ tok_slot) {
  int t = blockIdx.x * 256 + threadIdx.x;
  if (t >= Tv) return;
#pragma unroll
  for (int k = 0; k < Kv; ++k) {
    int e    = tok_e[t * 2 + k];
    int pos  = atomicAdd(&ctrl[8 + e], 1);
    int slot = ctrl[16 + e] + pos;
    slot_tok[slot]       = t;
    tok_slot[t * 2 + k]  = slot;
  }
}

// ---------------- WMMA fragment loads (ISA 7.12.2 bf16 layouts) ----------------
// A 16x32: lane row M=l&15; elems 0..7 -> K = hi*8+0..7, elems 8..15 -> K = 16+hi*8+0..7
__device__ __forceinline__ v16bf load_a_frag(const bf16_t* __restrict__ base, int k0, int hi) {
  v8bf lo8 = *(const v8bf*)(base + k0 + hi * 8);
  v8bf hi8 = *(const v8bf*)(base + k0 + 16 + hi * 8);
  v16bf r;
#pragma unroll
  for (int i = 0; i < 8; ++i) { r[i] = lo8[i]; r[i + 8] = hi8[i]; }
  return r;
}
// B 32x16: lane col N=l&15; elems 0..7 -> K = hi*8+0..7, elems 8..15 -> K = 16+hi*8+0..7
__device__ __forceinline__ v16bf load_b_frag(const bf16_t* __restrict__ base, int k0, int hi) {
  v8bf lo8 = *(const v8bf*)(base + k0 + hi * 8);
  v8bf hi8 = *(const v8bf*)(base + k0 + 16 + hi * 8);
  v16bf r;
#pragma unroll
  for (int i = 0; i < 8; ++i) { r[i] = lo8[i]; r[i + 8] = hi8[i]; }
  return r;
}

// ---------------- GEMM1: h = relu(gather(x) @ w1[e]^T + b1[e]) (bf16 out) ----------------
// block = 128 threads = 4 waves; block tile 128x128; wave tile 64x64 (4x4 WMMA accums)
__global__ __launch_bounds__(128) void gemm1_kernel(
    const bf16_t* __restrict__ xb,   // [Tv, Cv] bf16
    const bf16_t* __restrict__ w1b,  // [Ev, Fv, Cv] bf16
    const float*  __restrict__ bias1,// [Ev, Fv]
    const int*    __restrict__ ctrl,
    const int*    __restrict__ slot_tok,
    bf16_t*       __restrict__ h) {  // [SLOT_MAX, Fv] bf16
  const int e     = blockIdx.z;
  const int cnt   = ctrl[e];
  const int mtile = blockIdx.y;
  if (mtile * TILE_M >= cnt) return;                   // uniform early exit
  const int off   = ctrl[16 + e];
  const int lane  = threadIdx.x & 31;
  const int wave  = threadIdx.x >> 5;
  const int lo    = lane & 15, hi = lane >> 4;
  const int mbase = mtile * TILE_M + (wave >> 1) * 64; // local row base in expert region
  const int nbase = blockIdx.x * TILE_M + (wave & 1) * 64;

  const bf16_t* aptr[4];
#pragma unroll
  for (int i = 0; i < 4; ++i) {
    int lrow = mbase + i * 16 + lo;
    int tok  = (lrow < cnt) ? slot_tok[off + lrow] : 0;  // clamp pad rows to token 0
    aptr[i]  = xb + (size_t)tok * Cv;
  }
  const bf16_t* bptr[4];
#pragma unroll
  for (int j = 0; j < 4; ++j)
    bptr[j] = w1b + ((size_t)e * Fv + (size_t)(nbase + j * 16 + lo)) * Cv;

  v8f acc[4][4];
#pragma unroll
  for (int i = 0; i < 4; ++i)
#pragma unroll
    for (int j = 0; j < 4; ++j)
#pragma unroll
      for (int r = 0; r < 8; ++r) acc[i][j][r] = 0.f;

  for (int k0 = 0; k0 < Cv; k0 += 32) {
    v16bf a[4], b[4];
#pragma unroll
    for (int i = 0; i < 4; ++i) a[i] = load_a_frag(aptr[i], k0, hi);
#pragma unroll
    for (int j = 0; j < 4; ++j) b[j] = load_b_frag(bptr[j], k0, hi);
#pragma unroll
    for (int i = 0; i < 4; ++i)
#pragma unroll
      for (int j = 0; j < 4; ++j)
        acc[i][j] = __builtin_amdgcn_wmma_f32_16x16x32_bf16(
            false, a[i], false, b[j], (short)0, acc[i][j], false, false);
  }

  // epilogue: +bias, relu, bf16 store. D layout: elem r -> M = r + 8*hi, N = l&15
#pragma unroll
  for (int j = 0; j < 4; ++j) {
    int   col = nbase + j * 16 + lo;
    float bv  = bias1[e * Fv + col];
#pragma unroll
    for (int i = 0; i < 4; ++i) {
      int rowb = off + mbase + i * 16 + 8 * hi;
#pragma unroll
      for (int r = 0; r < 8; ++r) {
        float v = acc[i][j][r] + bv;
        v = v > 0.f ? v : 0.f;
        h[(size_t)(rowb + r) * Fv + col] = (bf16_t)v;
      }
    }
  }
}

// ---------------- GEMM2: y = h @ w2[e]^T + b2[e] (fp32 out, per slot) ----------------
__global__ __launch_bounds__(128) void gemm2_kernel(
    const bf16_t* __restrict__ hb,   // [SLOT_MAX, Fv] bf16
    const bf16_t* __restrict__ w2b,  // [Ev, Cv, Fv] bf16
    const float*  __restrict__ bias2,// [Ev, Cv]
    const int*    __restrict__ ctrl,
    float*        __restrict__ y) {  // [SLOT_MAX, Cv] fp32
  const int e     = blockIdx.z;
  const int cnt   = ctrl[e];
  const int mtile = blockIdx.y;
  if (mtile * TILE_M >= cnt) return;
  const int off   = ctrl[16 + e];
  const int lane  = threadIdx.x & 31;
  const int wave  = threadIdx.x >> 5;
  const int lo    = lane & 15, hi = lane >> 4;
  const int mbase = mtile * TILE_M + (wave >> 1) * 64;
  const int nbase = blockIdx.x * TILE_M + (wave & 1) * 64;

  const bf16_t* aptr[4];
#pragma unroll
  for (int i = 0; i < 4; ++i)
    aptr[i] = hb + (size_t)(off + mbase + i * 16 + lo) * Fv;   // rows contiguous, no gather
  const bf16_t* bptr[4];
#pragma unroll
  for (int j = 0; j < 4; ++j)
    bptr[j] = w2b + ((size_t)e * Cv + (size_t)(nbase + j * 16 + lo)) * Fv;

  v8f acc[4][4];
#pragma unroll
  for (int i = 0; i < 4; ++i)
#pragma unroll
    for (int j = 0; j < 4; ++j)
#pragma unroll
      for (int r = 0; r < 8; ++r) acc[i][j][r] = 0.f;

  for (int k0 = 0; k0 < Fv; k0 += 32) {
    v16bf a[4], b[4];
#pragma unroll
    for (int i = 0; i < 4; ++i) a[i] = load_a_frag(aptr[i], k0, hi);
#pragma unroll
    for (int j = 0; j < 4; ++j) b[j] = load_b_frag(bptr[j], k0, hi);
#pragma unroll
    for (int i = 0; i < 4; ++i)
#pragma unroll
      for (int j = 0; j < 4; ++j)
        acc[i][j] = __builtin_amdgcn_wmma_f32_16x16x32_bf16(
            false, a[i], false, b[j], (short)0, acc[i][j], false, false);
  }

#pragma unroll
  for (int j = 0; j < 4; ++j) {
    int   col = nbase + j * 16 + lo;
    float bv  = bias2[e * Cv + col];
#pragma unroll
    for (int i = 0; i < 4; ++i) {
      int rowb = off + mbase + i * 16 + 8 * hi;
#pragma unroll
      for (int r = 0; r < 8; ++r)
        y[(size_t)(rowb + r) * Cv + col] = acc[i][j][r] + bv;
    }
  }
}

// ---------------- combine: out[t] = w0*y[slot0] + w1*y[slot1] ----------------
__global__ __launch_bounds__(256) void combine_kernel(
    const float* __restrict__ y, const int* __restrict__ tok_slot,
    const float* __restrict__ tok_w, float* __restrict__ out) {
  int idx = blockIdx.x * 256 + threadIdx.x;            // over Tv*Cv/4
  int t   = idx / (Cv / 4);
  int c   = (idx % (Cv / 4)) * 4;
  if (t >= Tv) return;
  int   s0 = tok_slot[t * 2], s1 = tok_slot[t * 2 + 1];
  float w0 = tok_w[t * 2],    w1 = tok_w[t * 2 + 1];
  v4f y0 = *(const v4f*)(y + (size_t)s0 * Cv + c);
  v4f y1 = *(const v4f*)(y + (size_t)s1 * Cv + c);
  v4f o;
#pragma unroll
  for (int r = 0; r < 4; ++r) o[r] = w0 * y0[r] + w1 * y1[r];
  *(v4f*)(out + (size_t)t * Cv + c) = o;
}

// ---------------- host launch ----------------
extern "C" void kernel_launch(void* const* d_in, const int* in_sizes, int n_in,
                              void* d_out, int out_size, void* d_ws, size_t ws_size,
                              hipStream_t stream) {
  (void)in_sizes; (void)n_in; (void)out_size; (void)ws_size;
  const float* x  = (const float*)d_in[0];   // [B,N,C]
  const float* rw = (const float*)d_in[1];   // [E,C]
  const float* w1 = (const float*)d_in[2];   // [E,F,C]
  const float* b1 = (const float*)d_in[3];   // [E,F]
  const float* w2 = (const float*)d_in[4];   // [E,C,F]
  const float* b2 = (const float*)d_in[5];   // [E,C]
  float* out = (float*)d_out;

  // carve workspace (~366 MB total)
  char*  ws = (char*)d_ws;
  size_t p  = 0;
  auto take = [&](size_t bytes) -> char* {
    char* cur = ws + p;
    p = (p + bytes + 255) & ~(size_t)255;
    return cur;
  };
  int*    ctrl     = (int*)take(32 * sizeof(int));
  int*    tok_e    = (int*)take((size_t)Tv * 2 * sizeof(int));
  float*  tok_w    = (float*)take((size_t)Tv * 2 * sizeof(float));
  int*    tok_slot = (int*)take((size_t)Tv * 2 * sizeof(int));
  int*    slot_tok = (int*)take((size_t)SLOT_MAX * sizeof(int));
  bf16_t* x_bf     = (bf16_t*)take((size_t)Tv * Cv * 2);
  bf16_t* w1_bf    = (bf16_t*)take((size_t)Ev * Fv * Cv * 2);
  bf16_t* w2_bf    = (bf16_t*)take((size_t)Ev * Cv * Fv * 2);
  bf16_t* h_bf     = (bf16_t*)take((size_t)SLOT_MAX * Fv * 2);
  float*  y_buf    = (float*)take((size_t)SLOT_MAX * Cv * sizeof(float));

  zero_ctrl_kernel<<<1, 32, 0, stream>>>(ctrl);

  const int nx  = Tv * Cv / 4;
  const int nw1 = Ev * Fv * Cv / 4;
  const int nw2 = Ev * Cv * Fv / 4;
  cast_bf16_kernel<<<(nx  + 255) / 256, 256, 0, stream>>>(x,  x_bf,  nx);
  cast_bf16_kernel<<<(nw1 + 255) / 256, 256, 0, stream>>>(w1, w1_bf, nw1);
  cast_bf16_kernel<<<(nw2 + 255) / 256, 256, 0, stream>>>(w2, w2_bf, nw2);

  router_kernel<<<Tv / 256, 256, 0, stream>>>(x, rw, ctrl, tok_e, tok_w);
  scan_kernel<<<1, 1, 0, stream>>>(ctrl);
  scatter_kernel<<<Tv / 256, 256, 0, stream>>>(tok_e, ctrl, slot_tok, tok_slot);

  gemm1_kernel<<<dim3(Fv / TILE_M, Tv / TILE_M, Ev), 128, 0, stream>>>(
      x_bf, w1_bf, b1, ctrl, slot_tok, h_bf);
  gemm2_kernel<<<dim3(Cv / TILE_M, Tv / TILE_M, Ev), 128, 0, stream>>>(
      h_bf, w2_bf, b2, ctrl, y_buf);

  combine_kernel<<<(Tv * Cv / 4) / 256, 256, 0, stream>>>(y_buf, tok_slot, tok_w, out);
}
